// IrregularityAwareAttention_24275155157346
// MI455X (gfx1250) — matrix operand order repaired and checked by reading, not compiled, not run on hardware
//
#include <hip/hip_runtime.h>

// ---------------------------------------------------------------------------
// IrregularityAwareAttention for MI455X (gfx1250, wave32, WMMA).
// B=8, N=2048, D_MODEL=512, D_AUX=64.
// Strategy: bf16 operands + f32 WMMA accumulation (v_wmma_f32_16x16x32_bf16),
// flash-attention style online softmax, adjacency mask + data-dependent merge
// applied in f32 between the two GEMMs. K-tiles staged into LDS via the
// Tensor Data Mover (tensor_load_to_lds + s_wait_tensorcnt) when available.
// ---------------------------------------------------------------------------

#define BATCH 8
#define SEQ   2048
#define DM    512
#define DAUX  64

typedef __attribute__((ext_vector_type(4)))  unsigned int u32x4;
typedef __attribute__((ext_vector_type(8)))  float        f32x8;
typedef __attribute__((ext_vector_type(16))) __bf16       bf16x16;

#if defined(__gfx1250__) && __has_builtin(__builtin_amdgcn_tensor_load_to_lds) && \
    __has_builtin(__builtin_amdgcn_s_wait_tensorcnt)
#define USE_TDM 1
typedef __attribute__((ext_vector_type(4))) unsigned int tdm_g0_t;
typedef __attribute__((ext_vector_type(8))) int          tdm_g1_t;
typedef __attribute__((ext_vector_type(4))) int          tdm_g2_t;
#else
#define USE_TDM 0
#endif

union Frag {
    u32x4   q[2];   // 32 bytes = 16 bf16
    bf16x16 v;
};

__device__ __forceinline__ f32x8 wmma_bf16(bf16x16 a, bf16x16 b, f32x8 c) {
    // D = A(16x32) * B(32x16) + C(16x16), f32 accumulate
    return __builtin_amdgcn_wmma_f32_16x16x32_bf16(
        /*neg_a=*/false, a, /*neg_b=*/false, b,
        /*c_mod=*/(short)0, c, /*reuse_a=*/false, /*reuse_b=*/false);
}

__device__ __forceinline__ unsigned short f32_to_bf16(float f) {
    unsigned int u = __float_as_uint(f);
    unsigned int r = (u + 0x7FFFu + ((u >> 16) & 1u)) >> 16;  // RNE
    return (unsigned short)r;
}

__device__ __forceinline__ f32x8 zero8() {
    f32x8 z = {0.f, 0.f, 0.f, 0.f, 0.f, 0.f, 0.f, 0.f};
    return z;
}

// ---------------------------------------------------------------------------
// Kernel 0: f32 -> bf16 conversion (elementwise).
// ---------------------------------------------------------------------------
__global__ __launch_bounds__(256) void cvt_f32_bf16_kernel(
    const float* __restrict__ src, unsigned short* __restrict__ dst, int n) {
    int i = blockIdx.x * 256 + threadIdx.x;
    if (i < n) dst[i] = f32_to_bf16(src[i]);
}

// ---------------------------------------------------------------------------
// Kernel 1: QKV projection.  y = x @ W^T + b, stored as bf16.
// Block = 256 threads (8 waves).  Block tile = 32 rows x 64 cols,
// wave tile = 16x16, K loop = 16 steps of 32 (bf16 WMMA).
// grid = (M/32, 512/64, 3); z selects {q,k,v}.
// ---------------------------------------------------------------------------
__global__ __launch_bounds__(256) void qkv_proj_kernel(
    const unsigned short* __restrict__ xb,   // [B*N, DM] bf16
    const unsigned short* __restrict__ Wqb,
    const unsigned short* __restrict__ Wkb,
    const unsigned short* __restrict__ Wvb,  // [DM, DM] bf16 (row = out feature)
    const float* __restrict__ bq,
    const float* __restrict__ bk,
    const float* __restrict__ bv,
    unsigned short* __restrict__ qb,
    unsigned short* __restrict__ kb,
    unsigned short* __restrict__ vb) {

    const int wid  = threadIdx.x >> 5;
    const int lane = threadIdx.x & 31;
    const int lm   = lane & 15;
    const int lh   = lane >> 4;

    const int m0 = blockIdx.x * 32 + (wid >> 2) * 16;
    const int n0 = blockIdx.y * 64 + (wid & 3) * 16;

    const unsigned short* W;
    const float*          bias;
    unsigned short*       out;
    if (blockIdx.z == 0)      { W = Wqb; bias = bq; out = qb; }
    else if (blockIdx.z == 1) { W = Wkb; bias = bk; out = kb; }
    else                      { W = Wvb; bias = bv; out = vb; }

    const unsigned short* arow = xb + (size_t)(m0 + lm) * DM;
    const unsigned short* brow = W  + (size_t)(n0 + lm) * DM;

    f32x8 acc = zero8();
#pragma unroll
    for (int ks = 0; ks < DM; ks += 32) {
        Frag a, b;
        // A 16x32 bf16: lane<16 -> K{0..7,16..23}, lane>=16 -> K{8..15,24..31}
        const unsigned short* ap = arow + ks + lh * 8;
        a.q[0] = *(const u32x4*)(ap);
        a.q[1] = *(const u32x4*)(ap + 16);
        // B 32x16 bf16: lane<16 -> K{0..7,16..23}, lane>=16 -> K{8..15,24..31}, col = lane%16
        const unsigned short* bp = brow + ks + lh * 8;
        b.q[0] = *(const u32x4*)(bp);
        b.q[1] = *(const u32x4*)(bp + 16);
        acc = wmma_bf16(a.v, b.v, acc);
    }

    const float bval = bias[n0 + lm];
#pragma unroll
    for (int r = 0; r < 8; ++r) {
        const int m = m0 + r + 8 * lh;  // C layout: row = vgpr + 8*(lane>=16)
        out[(size_t)m * DM + n0 + lm] = f32_to_bf16(acc[r] + bval);
    }
}

// ---------------------------------------------------------------------------
// Kernel 2: flash attention with merge + adjacency mask.
// grid = (N/32, B); block = 256 threads = 8 waves.
// 2 row-groups of 16 query rows; each group has 4 waves splitting D=512.
// Key tiles of 32; K tile row-major in LDS (TDM DMA when available),
// V tile transposed in LDS.
// ---------------------------------------------------------------------------
#define LDS_K   0                       // 32 x 512 bf16 = 32768 B
#define LDS_VT  32768                   // 512 x 32 bf16 = 32768 B
#define LDS_SP  65536                   // 2 x 4 x 512 f32 = 16384 B
#define LDS_P   81920                   // 2 x 16 x 32 bf16 = 2048 B
#define LDS_TOTAL 83968

__global__ __launch_bounds__(256) void flash_attn_kernel(
    const unsigned short* __restrict__ qb,
    const unsigned short* __restrict__ kb,
    const unsigned short* __restrict__ vb,     // [B*N, DM] bf16
    const unsigned short* __restrict__ qauxb,
    const unsigned short* __restrict__ kauxb,  // [B*N, DAUX] bf16
    const int*            __restrict__ adj,    // [B, N, N] int32
    const float*          __restrict__ merge_p,
    const float*          __restrict__ thr_p,
    float*                __restrict__ out) {  // [B, N, DM] f32

    extern __shared__ char lds[];
    unsigned short* Kt    = (unsigned short*)(lds + LDS_K);
    unsigned short* Vt    = (unsigned short*)(lds + LDS_VT);
    float*          Spart = (float*)(lds + LDS_SP);
    unsigned short* Pbuf  = (unsigned short*)(lds + LDS_P);

    const int tid  = threadIdx.x;
    const int wid  = tid >> 5;
    const int lane = tid & 31;
    const int g    = wid >> 2;   // row group 0/1
    const int ws   = wid & 3;    // D-slice / K-reduction slice 0..3
    const int lm   = lane & 15;
    const int lh   = lane >> 4;

    const int b     = blockIdx.y;
    const int qrow0 = blockIdx.x * 32 + g * 16;

    const float scale     = 0.044194173824159216f;  // 1/sqrt(512)
    const float scale_aux = 0.125f;                 // 1/sqrt(64)
    const float mc  = *merge_p;
    const float thr = *thr_p;
    const float NEG = -3.402823466e38f;             // finfo(f32).min, as ref

    // --- resident Q fragments: this wave's 128-wide D slice (4 k-steps) ---
    Frag qa[4];
    {
        const unsigned short* qrow = qb + (size_t)(b * SEQ + qrow0 + lm) * DM;
#pragma unroll
        for (int s = 0; s < 4; ++s) {
            const int d = ws * 128 + s * 32 + lh * 8;
            qa[s].q[0] = *(const u32x4*)(qrow + d);
            qa[s].q[1] = *(const u32x4*)(qrow + d + 16);
        }
    }
    // --- resident aux-Q fragments (D_AUX=64 -> 2 k-steps) ---
    Frag qx[2];
    {
        const unsigned short* qarow = qauxb + (size_t)(b * SEQ + qrow0 + lm) * DAUX;
#pragma unroll
        for (int s = 0; s < 2; ++s) {
            const int d = s * 32 + lh * 8;
            qx[s].q[0] = *(const u32x4*)(qarow + d);
            qx[s].q[1] = *(const u32x4*)(qarow + d + 16);
        }
    }

    f32x8 o[8];
#pragma unroll
    for (int t = 0; t < 8; ++t) o[t] = zero8();
    float m_i[8], l_i[8];
#pragma unroll
    for (int r = 0; r < 8; ++r) { m_i[r] = NEG; l_i[r] = 0.f; }

    const unsigned short* Ksrc = kb + (size_t)(b * SEQ) * DM;
    const unsigned short* Vsrc = vb + (size_t)(b * SEQ) * DM;
    const int* adjbase = adj + ((size_t)b * SEQ + qrow0) * SEQ;

    for (int j = 0; j < SEQ / 32; ++j) {
        // ---- K tile (32 x 512 bf16, contiguous 32KB): TDM DMA into LDS ----
#if USE_TDM
        if (wid == 0) {
            const unsigned long long gaddr =
                (unsigned long long)(uintptr_t)(Ksrc + (size_t)j * 32 * DM);
            const unsigned ldsoff = (unsigned)(uintptr_t)Kt;
            // D# group0: count=1 (valid user descriptor), lds_addr,
            // 57-bit global_addr, type=2 ("image")
            tdm_g0_t g0 = { 1u,
                            ldsoff,
                            (unsigned)(gaddr & 0xffffffffu),
                            (unsigned)((gaddr >> 32) & 0x01ffffffu) | (2u << 30) };
            // D# group1: data_size=1 (2B elems); tensor_dim0 = tile_dim0 =
            // 16384 elems (one contiguous 1D tile); tensor_dim1 = 1.
            tdm_g1_t g1 = { (int)(1u << 16),       // workgroup_mask=0, data_size=2B
                            (int)(16384u << 16),   // tensor_dim0[15:0] << 16
                            (int)(1u << 16),       // tensor_dim0[31:16]=0 | tensor_dim1=1
                            (int)(16384u << 16),   // tensor_dim1 hi=0 | tile_dim0=16384
                            0,                     // tile_dim1=0, tile_dim2=0
                            16384,                 // tensor_dim0_stride[31:0]
                            0, 0 };
            tdm_g2_t gz4 = { 0, 0, 0, 0 };         // groups 2/3 unused (<=2D)
            tdm_g1_t gz8 = { 0, 0, 0, 0, 0, 0, 0, 0 };
            __builtin_amdgcn_tensor_load_to_lds(g0, g1, gz4, gz4, gz8, 0);
            __builtin_amdgcn_s_wait_tensorcnt(0);  // TENSORcnt -> 0
        }
#else
        {
            const u32x4* src = (const u32x4*)(Ksrc + (size_t)j * 32 * DM);
            u32x4*       dst = (u32x4*)Kt;
            for (int i = tid; i < 2048; i += 256) dst[i] = src[i];
        }
#endif
        // ---- V tile transposed into LDS (manual; strided b16 stores) ----
        {
            const unsigned short* vsrcj = Vsrc + (size_t)j * 32 * DM;
            for (int i = tid; i < 2048; i += 256) {
                const int key = i >> 6;
                const int d0  = (i & 63) * 8;
                u32x4 val = *(const u32x4*)(vsrcj + key * DM + d0);
                const unsigned short* e = (const unsigned short*)&val;
#pragma unroll
                for (int t = 0; t < 8; ++t) Vt[(d0 + t) * 32 + key] = e[t];
            }
            // prefetch next iteration's K tile and adjacency block (gfx1250
            // global_prefetch_b8)
            if (j + 1 < SEQ / 32) {
                __builtin_prefetch(Ksrc + (size_t)(j + 1) * 32 * DM + tid * 64, 0, 1);
                __builtin_prefetch(adjbase + (size_t)(tid & 15) * SEQ + (j + 1) * 32, 0, 1);
            }
        }
        __syncthreads();

        // ---- partial S over this wave's 128-wide slice of D ----
        f32x8 s0 = zero8(), s1 = zero8();
#pragma unroll
        for (int s = 0; s < 4; ++s) {
            Frag b0, b1;
            const unsigned short* kp0 = Kt + lm * DM        + ws * 128 + s * 32 + lh * 8;
            const unsigned short* kp1 = Kt + (16 + lm) * DM + ws * 128 + s * 32 + lh * 8;
            b0.q[0] = *(const u32x4*)(kp0); b0.q[1] = *(const u32x4*)(kp0 + 16);
            b1.q[0] = *(const u32x4*)(kp1); b1.q[1] = *(const u32x4*)(kp1 + 16);
            s0 = wmma_bf16(qa[s].v, b0.v, s0);
            s1 = wmma_bf16(qa[s].v, b1.v, s1);
        }
        {
            float* sp = Spart + (size_t)(g * 4 + ws) * 512;
#pragma unroll
            for (int r = 0; r < 8; ++r) {
                sp[r * 32 + lane]       = s0[r];
                sp[256 + r * 32 + lane] = s1[r];
            }
        }
        __syncthreads();

        // ---- reduce partial S across the 4 waves of the row group ----
        f32x8 S0 = zero8(), S1 = zero8();
#pragma unroll
        for (int w = 0; w < 4; ++w) {
            const float* sp = Spart + (size_t)(g * 4 + w) * 512;
#pragma unroll
            for (int r = 0; r < 8; ++r) {
                S0[r] += sp[r * 32 + lane];
                S1[r] += sp[256 + r * 32 + lane];
            }
        }

        // ---- aux scores (K = 64, done fully per wave) ----
        f32x8 A0 = zero8(), A1 = zero8();
        {
            const unsigned short* ka = kauxb + (size_t)(b * SEQ + j * 32) * DAUX;
#pragma unroll
            for (int s = 0; s < 2; ++s) {
                Frag b0, b1;
                const unsigned short* p0 = ka + lm * DAUX        + s * 32 + lh * 8;
                const unsigned short* p1 = ka + (16 + lm) * DAUX + s * 32 + lh * 8;
                b0.q[0] = *(const u32x4*)(p0); b0.q[1] = *(const u32x4*)(p0 + 16);
                b1.q[0] = *(const u32x4*)(p1); b1.q[1] = *(const u32x4*)(p1 + 16);
                A0 = wmma_bf16(qx[s].v, b0.v, A0);
                A1 = wmma_bf16(qx[s].v, b1.v, A1);
            }
        }

        // ---- merge + mask + online softmax (C layout: row = r + 8*lh) ----
        const int* adjrow = adjbase + j * 32;
#pragma unroll
        for (int r = 0; r < 8; ++r) {
            const int row = r + 8 * lh;
            float sA = S0[r] * scale, sB = S1[r] * scale;
            const float xA = A0[r] * scale_aux, xB = A1[r] * scale_aux;
            if (xA != 0.f && sA > thr) sA = (1.f - mc) * sA + mc * xA;
            if (xB != 0.f && sB > thr) sB = (1.f - mc) * sB + mc * xB;
            if (adjrow[(size_t)row * SEQ + lm] == 0)      sA = NEG;
            if (adjrow[(size_t)row * SEQ + 16 + lm] == 0) sB = NEG;

            float mx = fmaxf(sA, sB);
            mx = fmaxf(mx, __shfl_xor(mx, 1, 32));
            mx = fmaxf(mx, __shfl_xor(mx, 2, 32));
            mx = fmaxf(mx, __shfl_xor(mx, 4, 32));
            mx = fmaxf(mx, __shfl_xor(mx, 8, 32));

            const float mN    = fmaxf(m_i[r], mx);
            const float alpha = __expf(m_i[r] - mN);
            m_i[r] = mN;

            const float pA = __expf(sA - mN);
            const float pB = __expf(sB - mN);
            float rs = pA + pB;
            rs += __shfl_xor(rs, 1, 32);
            rs += __shfl_xor(rs, 2, 32);
            rs += __shfl_xor(rs, 4, 32);
            rs += __shfl_xor(rs, 8, 32);
            l_i[r] = l_i[r] * alpha + rs;

#pragma unroll
            for (int t = 0; t < 8; ++t) o[t][r] *= alpha;
            S0[r] = pA;
            S1[r] = pB;
        }

        // ---- P (bf16) to LDS; all 4 waves of a group write identical data ----
        {
            unsigned short* pb = Pbuf + g * 512;
#pragma unroll
            for (int r = 0; r < 8; ++r) {
                const int row = r + 8 * lh;
                pb[row * 32 + lm]      = f32_to_bf16(S0[r]);
                pb[row * 32 + 16 + lm] = f32_to_bf16(S1[r]);
            }
        }
        __syncthreads();

        // ---- O += P (16x32) @ V (32x128 slice), 8 WMMAs ----
        {
            Frag pf;
            const unsigned short* pp = Pbuf + g * 512 + lm * 32 + lh * 8;
            pf.q[0] = *(const u32x4*)(pp);
            pf.q[1] = *(const u32x4*)(pp + 16);
#pragma unroll
            for (int t = 0; t < 8; ++t) {
                Frag vf;
                const int n = ws * 128 + t * 16 + lm;          // output col = Vt row
                const unsigned short* vp = Vt + n * 32 + lh * 8;
                vf.q[0] = *(const u32x4*)(vp);
                vf.q[1] = *(const u32x4*)(vp + 16);
                o[t] = wmma_bf16(pf.v, vf.v, o[t]);
            }
        }
        __syncthreads();  // protect K/V/P/Spart before next iteration's writes
    }

    // ---- epilogue: O / l ----
    float* outp = out + (size_t)(b * SEQ + qrow0) * DM;
#pragma unroll
    for (int r = 0; r < 8; ++r) {
        const int row = r + 8 * lh;
        const float inv = 1.f / l_i[r];
#pragma unroll
        for (int t = 0; t < 8; ++t) {
            outp[(size_t)row * DM + ws * 128 + t * 16 + lm] = o[t][r] * inv;
        }
    }
}

// ---------------------------------------------------------------------------
// Host launcher
// ---------------------------------------------------------------------------
extern "C" void kernel_launch(void* const* d_in, const int* in_sizes, int n_in,
                              void* d_out, int out_size, void* d_ws, size_t ws_size,
                              hipStream_t stream) {
    (void)in_sizes; (void)n_in; (void)out_size; (void)ws_size;

    const float* x     = (const float*)d_in[0];
    const float* qaux  = (const float*)d_in[1];
    const float* kaux  = (const float*)d_in[2];
    const int*   adjm  = (const int*)d_in[3];
    const float* mcoef = (const float*)d_in[4];
    const float* Wq    = (const float*)d_in[5];
    const float* bq    = (const float*)d_in[6];
    const float* Wk    = (const float*)d_in[7];
    const float* bk    = (const float*)d_in[8];
    const float* Wv    = (const float*)d_in[9];
    const float* bv    = (const float*)d_in[10];
    const float* thr   = (const float*)d_in[11];
    float*       outp  = (float*)d_out;

    // workspace layout (bf16 staging buffers), 256B aligned
    char* ws = (char*)d_ws;
    size_t off = 0;
    auto alloc = [&](size_t bytes) {
        size_t o = off;
        off += (bytes + 255) & ~(size_t)255;
        return o;
    };
    const size_t NX   = (size_t)BATCH * SEQ * DM;    // 8.39M elems
    const size_t NW   = (size_t)DM * DM;             // 262k elems
    const size_t NAUX = (size_t)BATCH * SEQ * DAUX;  // 1.05M elems

    unsigned short* xb    = (unsigned short*)(ws + alloc(NX * 2));
    unsigned short* Wqb   = (unsigned short*)(ws + alloc(NW * 2));
    unsigned short* Wkb   = (unsigned short*)(ws + alloc(NW * 2));
    unsigned short* Wvb   = (unsigned short*)(ws + alloc(NW * 2));
    unsigned short* qb    = (unsigned short*)(ws + alloc(NX * 2));
    unsigned short* kb    = (unsigned short*)(ws + alloc(NX * 2));
    unsigned short* vb    = (unsigned short*)(ws + alloc(NX * 2));
    unsigned short* qauxb = (unsigned short*)(ws + alloc(NAUX * 2));
    unsigned short* kauxb = (unsigned short*)(ws + alloc(NAUX * 2));

    auto cvt = [&](const float* s, unsigned short* d, size_t n) {
        cvt_f32_bf16_kernel<<<dim3((unsigned)((n + 255) / 256)), dim3(256), 0, stream>>>(
            s, d, (int)n);
    };
    cvt(x, xb, NX);
    cvt(Wq, Wqb, NW);
    cvt(Wk, Wkb, NW);
    cvt(Wv, Wvb, NW);
    cvt(qaux, qauxb, NAUX);
    cvt(kaux, kauxb, NAUX);

    // QKV projection: grid (M/32, 512/64, 3)
    qkv_proj_kernel<<<dim3(BATCH * SEQ / 32, DM / 64, 3), dim3(256), 0, stream>>>(
        xb, Wqb, Wkb, Wvb, bq, bk, bv, qb, kb, vb);

    // Flash attention: grid (N/32, B), 84KB dynamic LDS
    flash_attn_kernel<<<dim3(SEQ / 32, BATCH), dim3(256), LDS_TOTAL, stream>>>(
        qb, kb, vb, qauxb, kauxb, adjm, mcoef, thr, outp);
}
